// SVDAdapter_25469156065845
// MI455X (gfx1250) — matrix operand order repaired and checked by reading, not compiled, not run on hardware
//
#include <hip/hip_runtime.h>

// ---------------------------------------------------------------------------
// SVD/LoRA adapter: y = x @ (W + (U*s) V^T)^T
//   x: (4,2048,4096) f32 -> M=8192, K=4096 ; W: (4096,4096) ; rank 16 adapter.
//
// MI455X strategy:
//   * Compute-bound GEMM (275 GFLOP vs ~14us of HBM traffic) -> run it on
//     V_WMMA_F32_16X16X32_BF16 with split-bf16 operands (hi/lo), f32 accum:
//     a*b ~= a_hi*b_hi + a_hi*b_lo + a_lo*b_hi  (near-f32 accuracy,
//     3 WMMA issues per K=32 vs 8 issues of the K=4 f32 WMMA).
//   * f32 -> bf16 hi/lo conversion happens ONCE per element at LDS staging
//     (v_cvt_pk_bf16_f32 / v_fma_mix_f32_bf16), LDS holds bf16 tiles ->
//     halves LDS traffic and cuts conversion VALU ~3x vs per-wave conversion.
//   * Low-rank term folded in as one extra K-tile: Tpad=(x@V)*s and Upad=U,
//     both zero-padded to 32 columns, streamed through the same WMMA loop.
//   * Global loads for tile k+1 issued during tile k's WMMA phase.
// ---------------------------------------------------------------------------

typedef __bf16 v8bf  __attribute__((ext_vector_type(8)));
typedef __bf16 v16bf __attribute__((ext_vector_type(16)));
typedef float  v8f   __attribute__((ext_vector_type(8)));
typedef float  v4f   __attribute__((ext_vector_type(4)));

#define K_DIM 4096
#define N_DIM 4096
#define M_DIM 8192

#define BM 128
#define BN 64
#define BK 32
#define LDSA 40   // bf16 row stride: 80B -> all-16-rows distinct banks, 16B aligned

// ---------------------------------------------------------------------------
// prep_T: Tpad[m][r] = sum_k x[m][k]*V[k][r]*s[r] (r<16), zeros for r in 16..31
// ---------------------------------------------------------------------------
__global__ __launch_bounds__(256)
void svdlora_prep_T(const float* __restrict__ X, const float* __restrict__ V,
                    const float* __restrict__ s, float* __restrict__ Tpad) {
    __shared__ float sx[16 * 256];
    const int tid = threadIdx.x;
    const int m0  = blockIdx.x * 16;
    const int ml  = tid >> 4;
    const int r   = tid & 15;
    const int lr  = tid >> 4;
    const int lc  = (tid & 15) * 16;

    float a0 = 0.f, a1 = 0.f, a2 = 0.f, a3 = 0.f;
    for (int kt = 0; kt < K_DIM / 256; ++kt) {
        const int k0 = kt * 256;
        __syncthreads();
        {
            const v4f* g  = (const v4f*)(X + (size_t)(m0 + lr) * K_DIM + k0 + lc);
            v4f*       sp = (v4f*)&sx[lr * 256 + lc];
            sp[0] = g[0]; sp[1] = g[1]; sp[2] = g[2]; sp[3] = g[3];
        }
        __syncthreads();
        const float* xr = &sx[ml * 256];
#pragma unroll 8
        for (int kk = 0; kk < 256; kk += 4) {
            a0 = fmaf(xr[kk + 0], V[(size_t)(k0 + kk + 0) * 16 + r], a0);
            a1 = fmaf(xr[kk + 1], V[(size_t)(k0 + kk + 1) * 16 + r], a1);
            a2 = fmaf(xr[kk + 2], V[(size_t)(k0 + kk + 2) * 16 + r], a2);
            a3 = fmaf(xr[kk + 3], V[(size_t)(k0 + kk + 3) * 16 + r], a3);
        }
    }
    const float acc = (a0 + a1) + (a2 + a3);
    Tpad[(size_t)(m0 + ml) * 32 + r]      = acc * s[r];
    Tpad[(size_t)(m0 + ml) * 32 + 16 + r] = 0.0f;
}

// ---------------------------------------------------------------------------
// prep_U: Upad[n][0..15] = U[n][0..15], Upad[n][16..31] = 0
// ---------------------------------------------------------------------------
__global__ __launch_bounds__(256)
void svdlora_prep_U(const float* __restrict__ U, float* __restrict__ Upad) {
    const int n = blockIdx.x * 256 + threadIdx.x;
    const v4f* u  = (const v4f*)(U + (size_t)n * 16);
    v4f*       up = (v4f*)(Upad + (size_t)n * 32);
    up[0] = u[0]; up[1] = u[1]; up[2] = u[2]; up[3] = u[3];
    v4f z = {0.f, 0.f, 0.f, 0.f};
    up[4] = z; up[5] = z; up[6] = z; up[7] = z;
}

// ---------------------------------------------------------------------------
// Main GEMM. 256 threads = 8 waves; block tile 128x64; wave tile 32x32
// (2x2 WMMA 16x16). LDS holds bf16 hi/lo tiles, converted at staging.
// ---------------------------------------------------------------------------
__device__ __forceinline__ v16bf frag_cat(v8bf x, v8bf y) {
    return __builtin_shufflevector(x, y, 0, 1, 2, 3, 4, 5, 6, 7,
                                         8, 9, 10, 11, 12, 13, 14, 15);
}

__global__ __launch_bounds__(256)
void svdlora_gemm(const float* __restrict__ X, const float* __restrict__ W,
                  const float* __restrict__ Tpad, const float* __restrict__ Upad,
                  float* __restrict__ Y) {
    __shared__ __bf16 sAhi[BM * LDSA];
    __shared__ __bf16 sAlo[BM * LDSA];
    __shared__ __bf16 sBhi[BN * LDSA];
    __shared__ __bf16 sBlo[BN * LDSA];

    const int tid  = threadIdx.x;
    const int wave = tid >> 5;
    const int lane = tid & 31;
    const int wm   = wave & 3;      // 32-row strip in 128
    const int wn   = wave >> 2;     // 32-col strip in 64
    const int l15  = lane & 15;
    const int lhi  = lane >> 4;

    const int m_blk = blockIdx.y * BM;
    const int n_blk = blockIdx.x * BN;

    // cooperative staging map: A -> 16 floats/thread, B -> 8 floats/thread
    const int rowA = tid >> 1;           // 0..127
    const int colA = (tid & 1) * 16;     // 0 | 16
    const int rowB = tid >> 2;           // 0..63
    const int colB = (tid & 3) * 8;      // 0,8,16,24

    const int NUM_KT = K_DIM / BK;       // 128 dense tiles + 1 low-rank tile

    auto tile_ptrs = [&](int t, const float*& aG, const float*& bG) {
        if (t < NUM_KT) {
            aG = X + (size_t)(m_blk + rowA) * K_DIM + t * BK + colA;
            bG = W + (size_t)(n_blk + rowB) * K_DIM + t * BK + colB;
        } else {
            aG = Tpad + (size_t)(m_blk + rowA) * 32 + colA;
            bG = Upad + (size_t)(n_blk + rowB) * 32 + colB;
        }
    };

    v8f acc[2][2] = {};

    // prologue: load tile 0 into registers
    v4f aR[4], bR[2];
    {
        const float *aG, *bG;
        tile_ptrs(0, aG, bG);
        const v4f* a4 = (const v4f*)aG;
        aR[0] = a4[0]; aR[1] = a4[1]; aR[2] = a4[2]; aR[3] = a4[3];
        const v4f* b4 = (const v4f*)bG;
        bR[0] = b4[0]; bR[1] = b4[1];
    }

    for (int kt = 0; kt <= NUM_KT; ++kt) {
        __syncthreads();   // previous tile's fragment reads are done

        // ---- convert registers (tile kt) to bf16 hi/lo, store to LDS ----
        {
            v8bf ah[2], al[2];
#pragma unroll
            for (int q = 0; q < 4; ++q) {
#pragma unroll
                for (int e = 0; e < 4; ++e) {
                    const int i = q * 4 + e;
                    const float v = aR[q][e];
                    const __bf16 h = (__bf16)v;
                    ah[i >> 3][i & 7] = h;
                    al[i >> 3][i & 7] = (__bf16)(v - (float)h);
                }
            }
            *(v8bf*)&sAhi[rowA * LDSA + colA]     = ah[0];
            *(v8bf*)&sAhi[rowA * LDSA + colA + 8] = ah[1];
            *(v8bf*)&sAlo[rowA * LDSA + colA]     = al[0];
            *(v8bf*)&sAlo[rowA * LDSA + colA + 8] = al[1];

            v8bf bh, bl;
#pragma unroll
            for (int q = 0; q < 2; ++q) {
#pragma unroll
                for (int e = 0; e < 4; ++e) {
                    const int i = q * 4 + e;
                    const float v = bR[q][e];
                    const __bf16 h = (__bf16)v;
                    bh[i] = h;
                    bl[i] = (__bf16)(v - (float)h);
                }
            }
            *(v8bf*)&sBhi[rowB * LDSA + colB] = bh;
            *(v8bf*)&sBlo[rowB * LDSA + colB] = bl;
        }
        __syncthreads();

        // ---- issue global loads for tile kt+1 (overlap with WMMA below) ----
        if (kt < NUM_KT) {
            const float *aG, *bG;
            tile_ptrs(kt + 1, aG, bG);
            const v4f* a4 = (const v4f*)aG;
            aR[0] = a4[0]; aR[1] = a4[1]; aR[2] = a4[2]; aR[3] = a4[3];
            const v4f* b4 = (const v4f*)bG;
            bR[0] = b4[0]; bR[1] = b4[1];
        }

        // ---- fragment reads (ds_load_b128 pairs) ----
        // A 16x32 bf16 layout: lane group 0 holds K 0..7 / 16..23,
        // lane group 1 holds K 8..15 / 24..31 of row M=l15.
        v16bf Ahi[2], Alo[2], Bhi[2], Blo[2];
#pragma unroll
        for (int mi = 0; mi < 2; ++mi) {
            const int off = (wm * 32 + mi * 16 + l15) * LDSA + lhi * 8;
            Ahi[mi] = frag_cat(*(const v8bf*)&sAhi[off], *(const v8bf*)&sAhi[off + 16]);
            Alo[mi] = frag_cat(*(const v8bf*)&sAlo[off], *(const v8bf*)&sAlo[off + 16]);
        }
        // B 32x16 bf16 layout: lane l15 = column, half-wave 0 holds K 0..7 / 16..23, half-wave 1 holds K 8..15 / 24..31.
#pragma unroll
        for (int ni = 0; ni < 2; ++ni) {
            const int off = (wn * 32 + ni * 16 + l15) * LDSA + lhi * 8;
            Bhi[ni] = frag_cat(*(const v8bf*)&sBhi[off], *(const v8bf*)&sBhi[off + 16]);
            Blo[ni] = frag_cat(*(const v8bf*)&sBlo[off], *(const v8bf*)&sBlo[off + 16]);
        }

        // ---- split-bf16 WMMA: hi*hi + hi*lo + lo*hi ----
#pragma unroll
        for (int mi = 0; mi < 2; ++mi) {
#pragma unroll
            for (int ni = 0; ni < 2; ++ni) {
                v8f c = acc[mi][ni];
                c = __builtin_amdgcn_wmma_f32_16x16x32_bf16(
                        false, Ahi[mi], false, Bhi[ni], (short)0, c, false, false);
                c = __builtin_amdgcn_wmma_f32_16x16x32_bf16(
                        false, Ahi[mi], false, Blo[ni], (short)0, c, false, false);
                c = __builtin_amdgcn_wmma_f32_16x16x32_bf16(
                        false, Alo[mi], false, Bhi[ni], (short)0, c, false, false);
                acc[mi][ni] = c;
            }
        }
    }

    // ---- epilogue: C layout -> VGPR j holds row (j + 8*half), col = l15 ----
#pragma unroll
    for (int mi = 0; mi < 2; ++mi) {
        const int rbase = m_blk + wm * 32 + mi * 16 + lhi * 8;
#pragma unroll
        for (int ni = 0; ni < 2; ++ni) {
            const int col = n_blk + wn * 32 + ni * 16 + l15;
#pragma unroll
            for (int j = 0; j < 8; ++j) {
                Y[(size_t)(rbase + j) * N_DIM + col] = acc[mi][ni][j];
            }
        }
    }
}

// ---------------------------------------------------------------------------
extern "C" void kernel_launch(void* const* d_in, const int* in_sizes, int n_in,
                              void* d_out, int out_size, void* d_ws, size_t ws_size,
                              hipStream_t stream) {
    (void)in_sizes; (void)n_in; (void)out_size; (void)ws_size;
    const float* X  = (const float*)d_in[0];   // (4,2048,4096)
    const float* Wb = (const float*)d_in[1];   // (4096,4096)
    const float* U  = (const float*)d_in[2];   // (4096,16)
    const float* s  = (const float*)d_in[3];   // (16,)
    const float* V  = (const float*)d_in[4];   // (4096,16)
    float* Y = (float*)d_out;

    float* Tpad = (float*)d_ws;                   // 8192*32 f32 = 1 MB
    float* Upad = Tpad + (size_t)M_DIM * 32;      // 4096*32 f32 = 512 KB

    svdlora_prep_T<<<M_DIM / 16, 256, 0, stream>>>(X, V, s, Tpad);
    svdlora_prep_U<<<N_DIM / 256, 256, 0, stream>>>(U, Upad);

    dim3 grid(N_DIM / BN, M_DIM / BM);   // (64, 64)
    svdlora_gemm<<<grid, 256, 0, stream>>>(X, Wb, Tpad, Upad, Y);
}
